// GNN_LSTM_Surrogate_43782896615689
// MI455X (gfx1250) — matrix operand order repaired and checked by reading, not compiled, not run on hardware
//
#include <hip/hip_runtime.h>
#include <hip/hip_fp16.h>

typedef _Float16 half_t;
typedef __attribute__((ext_vector_type(16))) _Float16 v16h;
typedef __attribute__((ext_vector_type(8)))  float    v8f;

constexpr int NB  = 16;     // batch
constexpr int NT  = 365;    // timesteps
constexpr int NP  = 10;     // producers
constexpr int NIJ = 8;      // injectors
constexpr int NEP = 90;     // p2p edges
constexpr int NEI = 80;     // i2p edges
constexpr int FPD = 10, FID = 8, FED = 10;
constexpr int NH  = 128;    // gnn hidden
constexpr int NLH = 256;    // lstm hidden
constexpr int NBT = NB * NT;            // 5840
constexpr int G4  = 4 * NLH;            // 1024 gates

// ---------------- WMMA helpers ----------------

__device__ __forceinline__ v8f wmma32(v16h a, v16h b, v8f c) {
  return __builtin_amdgcn_wmma_f32_16x16x32_f16(false, a, false, b, (short)0, c,
                                                false, false);
}

// 16-bit A-matrix 16x32 fragment: lane r=lane&15, hi=lane>>4.
// VGPR0..3 : K = hi*8 + 0..7 ; VGPR4..7 : K = 16 + hi*8 + 0..7
// => two 16B loads at uint4 indices hi and 2+hi of the row.
union AFrag { v16h v; uint4 q[2]; };
__device__ __forceinline__ v16h load_a16(const half_t* row, int hi) {
  AFrag u;
  const uint4* p = (const uint4*)row;
  u.q[0] = p[hi];
  u.q[1] = p[2 + hi];
  return u.v;
}

// B fragments are pre-swizzled so a lane's 16 halfs are contiguous.
__device__ __forceinline__ v16h load_b16(const half_t* bsw, int kt, int nt,
                                         int ntiles, int lane) {
  return *(const v16h*)(bsw + (((size_t)(kt * ntiles + nt) * 32 + lane) << 4));
}

__device__ __forceinline__ float sigf(float x) { return 1.f / (1.f + expf(-x)); }

// ---------------- weight swizzle: (K,N) f32 row-major -> WMMA B fragments ----
// B 32x16 f16 layout matches A: out[tile*512 + lane*16 + j] with
// K = hi*8 + j for j<8, K = 16 + hi*8 + (j-8) for j>=8.
__global__ void k_swz(const float* __restrict__ W, half_t* __restrict__ out,
                      int K, int N) {
  int idx = blockIdx.x * 256 + threadIdx.x;
  if (idx >= K * N) return;
  int j    = idx & 15;
  int lane = (idx >> 4) & 31;
  int tile = idx >> 9;
  int ntiles = N >> 4;
  int kt = tile / ntiles, nt = tile - kt * ntiles;
  int hi = lane >> 4, c = lane & 15;
  int kk = (j < 8) ? (hi * 8 + j) : (16 + hi * 8 + (j - 8));
  out[idx] = (half_t)W[(size_t)(kt * 32 + kk) * N + nt * 16 + c];
}

// ---------------- encoders: out = relu(X @ W + b), K small ----------------
__global__ void k_encode(const float* __restrict__ X, const float* __restrict__ W,
                         const float* __restrict__ b, half_t* __restrict__ out,
                         int K) {
  int row = blockIdx.x;
  int n   = threadIdx.x;                 // 0..127
  const float* x = X + (size_t)row * K;
  float acc = b[n];
  for (int k = 0; k < K; ++k) acc += x[k] * W[k * NH + n];
  acc = acc > 0.f ? acc : 0.f;
  out[(size_t)row * NH + n] = (half_t)acc;
}

// ---------------- f32 -> f16 convert ----------------
__global__ void k_f2h(const float* __restrict__ in, half_t* __restrict__ out, int n) {
  int i = blockIdx.x * 256 + threadIdx.x;
  if (i < n) out[i] = (half_t)in[i];
}

// ---------------- message GEMM (gather + WMMA + atomic scatter) -------------
// rows = bt*E + e ; A row = [nsrc[bt, src[e], :] (128) | ef[bt, e, :] (128)]
// out: agg[bt, dst[e], :] += relu(A @ Bsw + bias)
__global__ void k_msg(const half_t* __restrict__ nsrc, const half_t* __restrict__ ef,
                      const half_t* __restrict__ Bsw, const float* __restrict__ bias,
                      const int* __restrict__ src, const int* __restrict__ dst,
                      float* __restrict__ agg, int E, int S) {
  int lane  = threadIdx.x;
  int mtile = blockIdx.x;
  int r  = lane & 15, hi = lane >> 4;
  int row = mtile * 16 + r;
  int bt  = row / E;
  int e   = row - bt * E;
  const half_t* srow = nsrc + ((size_t)bt * S + src[e]) * NH;
  const half_t* erow = ef + (size_t)row * NH;

  v8f acc[8];
#pragma unroll
  for (int i = 0; i < 8; ++i) acc[i] = (v8f){0, 0, 0, 0, 0, 0, 0, 0};

#pragma unroll
  for (int ks = 0; ks < 8; ++ks) {
    const half_t* arow = (ks < 4) ? (srow + ks * 32) : (erow + (ks - 4) * 32);
    v16h a = load_a16(arow, hi);
    // Issue the full k-step's B loads before any WMMA so the compiler can
    // clause them and drain the 8 WMMAs behind a single s_wait.
    v16h bf[8];
#pragma unroll
    for (int nt = 0; nt < 8; ++nt) bf[nt] = load_b16(Bsw, ks, nt, 8, lane);
#pragma unroll
    for (int nt = 0; nt < 8; ++nt) acc[nt] = wmma32(a, bf[nt], acc[nt]);
  }

  int n0 = lane & 15;
  int dsti[8];
#pragma unroll
  for (int v = 0; v < 8; ++v) {
    int row2 = mtile * 16 + v + 8 * hi;
    int bt2  = row2 / E;
    int e2   = row2 - bt2 * E;
    dsti[v]  = bt2 * NP + dst[e2];
  }
#pragma unroll
  for (int nt = 0; nt < 8; ++nt) {
    float bb = bias[nt * 16 + n0];
#pragma unroll
    for (int v = 0; v < 8; ++v) {
      float val = acc[nt][v] + bb;
      val = val > 0.f ? val : 0.f;
      atomicAdd(&agg[(size_t)dsti[v] * NH + nt * 16 + n0], val);
    }
  }
}

// ---------------- node update GEMM: h' = relu([h|ap|ai] @ Wu + bu) ----------
__global__ void k_update(const half_t* __restrict__ hp, const half_t* __restrict__ ap,
                         const half_t* __restrict__ ai, const half_t* __restrict__ Bsw,
                         const float* __restrict__ bias, half_t* __restrict__ hout,
                         float* __restrict__ emb) {
  int lane = threadIdx.x, mtile = blockIdx.x;
  int r = lane & 15, hi = lane >> 4;
  size_t row = (size_t)mtile * 16 + r;

  v8f acc[8];
#pragma unroll
  for (int i = 0; i < 8; ++i) acc[i] = (v8f){0, 0, 0, 0, 0, 0, 0, 0};

#pragma unroll
  for (int ks = 0; ks < 12; ++ks) {
    const half_t* base = (ks < 4) ? hp : (ks < 8 ? ap : ai);
    v16h a = load_a16(base + row * NH + (ks & 3) * 32, hi);
    v16h bf[8];
#pragma unroll
    for (int nt = 0; nt < 8; ++nt) bf[nt] = load_b16(Bsw, ks, nt, 8, lane);
#pragma unroll
    for (int nt = 0; nt < 8; ++nt) acc[nt] = wmma32(a, bf[nt], acc[nt]);
  }

  int n0 = lane & 15;
#pragma unroll
  for (int nt = 0; nt < 8; ++nt) {
    float bb = bias[nt * 16 + n0];
#pragma unroll
    for (int v = 0; v < 8; ++v) {
      size_t row2 = (size_t)mtile * 16 + v + 8 * hi;
      float val = acc[nt][v] + bb;
      val = val > 0.f ? val : 0.f;
      hout[row2 * NH + nt * 16 + n0] = (half_t)val;
      if (emb) emb[row2 * NH + nt * 16 + n0] = val;
    }
  }
}

// ---------------- generic f16 GEMM + bias -> f32 (LSTM input projections) ---
__global__ void k_gemm(const half_t* __restrict__ A, const half_t* __restrict__ Bsw,
                       const float* __restrict__ bias, float* __restrict__ out,
                       int K, int N) {
  int lane = threadIdx.x, mtile = blockIdx.x, ng = blockIdx.y;  // 4 ntiles / wave
  int r = lane & 15, hi = lane >> 4;
  const half_t* arow = A + ((size_t)mtile * 16 + r) * K;
  int ntiles = N >> 4;
  int ksteps = K >> 5;

  v8f acc[4];
#pragma unroll
  for (int i = 0; i < 4; ++i) acc[i] = (v8f){0, 0, 0, 0, 0, 0, 0, 0};

  for (int ks = 0; ks < ksteps; ++ks) {
    __builtin_prefetch(arow + (ks + 1) * 32, 0, 3);  // global_prefetch_b8
    v16h a = load_a16(arow + ks * 32, hi);
    v16h bf[4];
#pragma unroll
    for (int i = 0; i < 4; ++i) bf[i] = load_b16(Bsw, ks, ng * 4 + i, ntiles, lane);
#pragma unroll
    for (int i = 0; i < 4; ++i) acc[i] = wmma32(a, bf[i], acc[i]);
  }

  int n0 = lane & 15;
#pragma unroll
  for (int i = 0; i < 4; ++i) {
    int n = (ng * 4 + i) * 16 + n0;
    float bb = bias[n];
#pragma unroll
    for (int v = 0; v < 8; ++v) {
      size_t row2 = (size_t)mtile * 16 + v + 8 * hi;
      out[row2 * N + n] = acc[i][v] + bb;
    }
  }
}

// ---------------- LSTM recurrence: one persistent workgroup, WMMA from LDS --
// xproj: (NB*NT, 1024) row = b*NT + t (gate preactivations sans h@Wh)
__global__ void k_lstm(const float* __restrict__ xproj, const half_t* __restrict__ Whsw,
                       half_t* __restrict__ hout16, float* __restrict__ hout32) {
  __shared__ half_t hsh[16 * NLH];    // 8 KB
  __shared__ float  csh[16 * NLH];    // 16 KB
  __shared__ half_t zsh[16 * G4];     // 32 KB
  int tid  = threadIdx.x;             // 256 threads = 8 waves
  int wave = tid >> 5, lane = tid & 31;
  int r = lane & 15, hi = lane >> 4;

  for (int i = tid; i < 16 * NLH; i += 256) { hsh[i] = (half_t)0.f; csh[i] = 0.f; }
  __syncthreads();

  for (int t = 0; t < NT; ++t) {
    // z = h @ Wh (16 x 1024); wave handles ntiles 8*wave..8*wave+7
    const half_t* arow = hsh + r * NLH;
#pragma unroll 2
    for (int nn = 0; nn < 8; ++nn) {
      int nt = wave * 8 + nn;
      // preload the whole K chain of B fragments for this ntile
      v16h bf[8];
#pragma unroll
      for (int ks = 0; ks < 8; ++ks) bf[ks] = load_b16(Whsw, ks, nt, 64, lane);
      v8f acc = (v8f){0, 0, 0, 0, 0, 0, 0, 0};
#pragma unroll
      for (int ks = 0; ks < 8; ++ks) {
        v16h a = load_a16(arow + ks * 32, hi);               // ds_load_b128
        acc = wmma32(a, bf[ks], acc);
      }
      int n = nt * 16 + (lane & 15);
#pragma unroll
      for (int v = 0; v < 8; ++v) {
        int m = v + 8 * hi;                                  // batch index
        zsh[m * G4 + n] = (half_t)(acc[v] + xproj[((size_t)m * NT + t) * G4 + n]);
      }
      if (t + 1 < NT)                                        // next timestep
        __builtin_prefetch(&xproj[((size_t)(8 * hi) * NT + t + 1) * G4 + n], 0, 3);
    }
    __syncthreads();

    // gates: thread owns column n = tid for all 16 batches
    int n = tid;
    for (int b = 0; b < 16; ++b) {
      float zi = (float)zsh[b * G4 + n];
      float zf = (float)zsh[b * G4 + NLH + n];
      float zg = (float)zsh[b * G4 + 2 * NLH + n];
      float zo = (float)zsh[b * G4 + 3 * NLH + n];
      float c = sigf(zf) * csh[b * NLH + n] + sigf(zi) * tanhf(zg);
      float h = sigf(zo) * tanhf(c);
      csh[b * NLH + n] = c;
      hsh[b * NLH + n] = (half_t)h;
      hout16[((size_t)b * NT + t) * NLH + n] = (half_t)h;
      hout32[((size_t)b * NT + t) * NLH + n] = h;
    }
    __syncthreads();
  }
}

// ---------------- decoder MLP + softplus ----------------
__global__ void k_decoder(const float* __restrict__ h2,
                          const float* __restrict__ Wd1, const float* __restrict__ bd1,
                          const float* __restrict__ a1p,
                          const float* __restrict__ Wd2, const float* __restrict__ bd2,
                          const float* __restrict__ a2p,
                          const float* __restrict__ Wd3, const float* __restrict__ bd3,
                          float* __restrict__ oil, float* __restrict__ water) {
  __shared__ float d1[64], d2[64];
  int row = blockIdx.x, j = threadIdx.x;       // 64 threads
  const float* x = h2 + (size_t)row * NLH;

  float acc = bd1[j];
  for (int k = 0; k < NLH; ++k) acc += x[k] * Wd1[k * 64 + j];
  float a1 = a1p[0];
  d1[j] = acc >= 0.f ? acc : a1 * acc;
  __syncthreads();

  acc = bd2[j];
  for (int k = 0; k < 64; ++k) acc += d1[k] * Wd2[k * 64 + j];
  float a2 = a2p[0];
  d2[j] = acc >= 0.f ? acc : a2 * acc;
  __syncthreads();

  if (j < 2 * NP) {
    acc = bd3[j];
    for (int k = 0; k < 64; ++k) acc += d2[k] * Wd3[k * 2 * NP + j];
    float sp = acc > 20.f ? acc : log1pf(expf(acc));
    int p = j >> 1;
    if (j & 1) water[(size_t)row * NP + p] = sp;
    else       oil[(size_t)row * NP + p]   = sp;
  }
}

// ---------------- host driver ----------------
extern "C" void kernel_launch(void* const* d_in, const int* in_sizes, int n_in,
                              void* d_out, int out_size, void* d_ws, size_t ws_size,
                              hipStream_t stream) {
  const float* pf  = (const float*)d_in[0];
  const float* jf  = (const float*)d_in[1];
  const float* epf = (const float*)d_in[2];
  const float* eif = (const float*)d_in[3];
  const int*   ip2p = (const int*)d_in[4];
  const int*   ii2p = (const int*)d_in[5];
  const float* Wpe = (const float*)d_in[6];  const float* bpe = (const float*)d_in[7];
  const float* Wie = (const float*)d_in[8];  const float* bie = (const float*)d_in[9];
  const float* Wep = (const float*)d_in[10]; const float* bep = (const float*)d_in[11];
  const float* Wei = (const float*)d_in[12]; const float* bei = (const float*)d_in[13];
  const float* Wmp = (const float*)d_in[14]; const float* bmp = (const float*)d_in[15];
  const float* Wmi = (const float*)d_in[16]; const float* bmi = (const float*)d_in[17];
  const float* Wu  = (const float*)d_in[18]; const float* bu  = (const float*)d_in[19];
  const float* Wx0 = (const float*)d_in[20]; const float* Wh0 = (const float*)d_in[21];
  const float* lb0 = (const float*)d_in[22];
  const float* Wx1 = (const float*)d_in[23]; const float* Wh1 = (const float*)d_in[24];
  const float* lb1 = (const float*)d_in[25];
  const float* Wd1 = (const float*)d_in[26]; const float* bd1 = (const float*)d_in[27];
  const float* a1  = (const float*)d_in[28];
  const float* Wd2 = (const float*)d_in[29]; const float* bd2 = (const float*)d_in[30];
  const float* a2  = (const float*)d_in[31];
  const float* Wd3 = (const float*)d_in[32]; const float* bd3 = (const float*)d_in[33];
  (void)in_sizes; (void)n_in; (void)out_size; (void)ws_size;

  char* wptr = (char*)d_ws;
  auto carve = [&](size_t bytes) -> void* {
    void* p = (void*)wptr;
    wptr += (bytes + 255) & ~size_t(255);
    return p;
  };

  const size_t NPH = (size_t)NBT * NP * NH;      // 7,475,200
  half_t* hp0    = (half_t*)carve(NPH * 2);
  half_t* hp1    = (half_t*)carve(NPH * 2);
  half_t* hi16   = (half_t*)carve((size_t)NBT * NIJ * NH * 2);
  half_t* ep16   = (half_t*)carve((size_t)NBT * NEP * NH * 2);
  half_t* ei16   = (half_t*)carve((size_t)NBT * NEI * NH * 2);
  float*  aggp   = (float*)carve(NPH * 4);
  float*  aggi   = (float*)carve(NPH * 4);
  half_t* aggp16 = (half_t*)carve(NPH * 2);
  half_t* aggi16 = (half_t*)carve(NPH * 2);
  float*  xproj  = (float*)carve((size_t)NBT * G4 * 4);
  half_t* h1_16  = (half_t*)carve((size_t)NBT * NLH * 2);
  float*  h1_32  = (float*)carve((size_t)NBT * NLH * 4);
  half_t* h2_16  = (half_t*)carve((size_t)NBT * NLH * 2);
  float*  h2_32  = (float*)carve((size_t)NBT * NLH * 4);
  half_t* WmpS   = (half_t*)carve((size_t)3 * 256 * 128 * 2);
  half_t* WmiS   = (half_t*)carve((size_t)3 * 256 * 128 * 2);
  half_t* WuS    = (half_t*)carve((size_t)3 * 384 * 128 * 2);
  half_t* Wx0S   = (half_t*)carve((size_t)1280 * 1024 * 2);
  half_t* Wh0S   = (half_t*)carve((size_t)256 * 1024 * 2);
  half_t* Wx1S   = (half_t*)carve((size_t)256 * 1024 * 2);
  half_t* Wh1S   = (half_t*)carve((size_t)256 * 1024 * 2);

  float* out   = (float*)d_out;
  float* oil   = out;
  float* water = out + (size_t)NBT * NP;
  float* emb   = out + (size_t)2 * NBT * NP;

  auto swz = [&](const float* W, half_t* O, int K, int N) {
    int n = K * N;
    k_swz<<<(n + 255) / 256, 256, 0, stream>>>(W, O, K, N);
  };
  // stacked-layer weights: treating K'=L*K keeps tile order layer-contiguous
  swz(Wmp, WmpS, 3 * 256, 128);
  swz(Wmi, WmiS, 3 * 256, 128);
  swz(Wu,  WuS,  3 * 384, 128);
  swz(Wx0, Wx0S, 1280, 1024);
  swz(Wh0, Wh0S, 256, 1024);
  swz(Wx1, Wx1S, 256, 1024);
  swz(Wh1, Wh1S, 256, 1024);

  // encoders
  k_encode<<<NBT * NP,  NH, 0, stream>>>(pf,  Wpe, bpe, hp0,  FPD);
  k_encode<<<NBT * NIJ, NH, 0, stream>>>(jf,  Wie, bie, hi16, FID);
  k_encode<<<NBT * NEP, NH, 0, stream>>>(epf, Wep, bep, ep16, FED);
  k_encode<<<NBT * NEI, NH, 0, stream>>>(eif, Wei, bei, ei16, FED);

  const int* sp = ip2p;       const int* dp = ip2p + NEP;
  const int* si = ii2p;       const int* di = ii2p + NEI;

  half_t* hcur = hp0;
  half_t* hnxt = hp1;
  for (int l = 0; l < 3; ++l) {
    hipMemsetAsync(aggp, 0, NPH * 4, stream);
    hipMemsetAsync(aggi, 0, NPH * 4, stream);
    k_msg<<<NBT * NEP / 16, 32, 0, stream>>>(hcur, ep16, WmpS + (size_t)l * 256 * 128,
                                             bmp + l * NH, sp, dp, aggp, NEP, NP);
    k_msg<<<NBT * NEI / 16, 32, 0, stream>>>(hi16, ei16, WmiS + (size_t)l * 256 * 128,
                                             bmi + l * NH, si, di, aggi, NEI, NIJ);
    int ncv = (int)NPH;
    k_f2h<<<(ncv + 255) / 256, 256, 0, stream>>>(aggp, aggp16, ncv);
    k_f2h<<<(ncv + 255) / 256, 256, 0, stream>>>(aggi, aggi16, ncv);
    k_update<<<NBT * NP / 16, 32, 0, stream>>>(hcur, aggp16, aggi16,
                                               WuS + (size_t)l * 384 * 128, bu + l * NH,
                                               hnxt, (l == 2) ? emb : nullptr);
    half_t* t = hcur; hcur = hnxt; hnxt = t;
  }

  // LSTM layer 0: x = emb flat (NBT, NP*NH=1280) == hcur rows
  k_gemm<<<dim3(NBT / 16, G4 / 64), 32, 0, stream>>>(hcur, Wx0S, lb0, xproj, 1280, G4);
  k_lstm<<<1, 256, 0, stream>>>(xproj, Wh0S, h1_16, h1_32);

  // LSTM layer 1
  k_gemm<<<dim3(NBT / 16, G4 / 64), 32, 0, stream>>>(h1_16, Wx1S, lb1, xproj, 256, G4);
  k_lstm<<<1, 256, 0, stream>>>(xproj, Wh1S, h2_16, h2_32);

  // decoder
  k_decoder<<<NBT, 64, 0, stream>>>(h2_32, Wd1, bd1, a1, Wd2, bd2, a2, Wd3, bd3,
                                    oil, water);
}
